// K_Means_44418551776003
// MI455X (gfx1250) — matrix operand order repaired and checked by reading, not compiled, not run on hardware
//
#include <hip/hip_runtime.h>
#include <cstdint>
#include <cstddef>

#define NPTS 262144
#define DIMS 256
#define NC   256
#define ROWS_PER_BLK 128
#define BLOCKS_ASSIGN (NPTS / ROWS_PER_BLK)   // 2048
#define CPAD 264                               // padded LDS stride (elements)

typedef float    v8f  __attribute__((ext_vector_type(8)));
typedef _Float16 v16h __attribute__((ext_vector_type(16)));

union Frag { _Float16 h[16]; uint4 q[2]; v16h v; };

// ---------------------------------------------------------------- csq
__global__ void csq_kernel(const float* __restrict__ cent, float* __restrict__ csq) {
  int k = threadIdx.x;
  float s = 0.f;
  for (int d = 0; d < DIMS; ++d) { float v = cent[k * DIMS + d]; s += v * v; }
  csq[k] = s;
}

// ---------------------------------------------------------------- assign (split-f16 WMMA GEMM + argmin)
__global__ __launch_bounds__(256, 1) void assign_kernel(
    const float* __restrict__ x, const float* __restrict__ cent,
    const float* __restrict__ csq, int* __restrict__ assign_i,
    float* __restrict__ assign_f, float* __restrict__ blk_part) {
  // centroids, f16 hi/lo split, dim-major (transposed), 2 x ~132KB
  __shared__ _Float16 ldsH[DIMS * CPAD];
  __shared__ _Float16 ldsL[DIMS * CPAD];
  __shared__ float lds_csq[NC];
  __shared__ float lds_xsq[ROWS_PER_BLK];
  __shared__ float lds_minv[ROWS_PER_BLK];

  const int tid = threadIdx.x;

  // Stage transposed split centroids: ldsH/L[d*CPAD + k] from cent[k][d].
  for (int d = 0; d < DIMS; ++d) {
    float f = cent[(size_t)tid * DIMS + d];
    _Float16 h = (_Float16)f;
    ldsH[d * CPAD + tid] = h;
    ldsL[d * CPAD + tid] = (_Float16)(f - (float)h);
  }
  lds_csq[tid] = csq[tid];
  __syncthreads();

  const int wave = tid >> 5, lane = tid & 31;
  const int half = lane >> 4, l15 = lane & 15;
  const int rowBase = blockIdx.x * ROWS_PER_BLK + wave * 16;
  const float* xrow = x + (size_t)(rowBase + l15) * DIMS;

  v8f acc[16];
#pragma unroll
  for (int t = 0; t < 16; ++t)
#pragma unroll
    for (int j = 0; j < 8; ++j) acc[t][j] = 0.f;

  float xsq = 0.f;
  for (int d0 = 0; d0 < DIMS; d0 += 32) {
    // A fragment: 16-bit A 16x32 layout — lane holds row (l15),
    // elements 0..7 = dims d0+half*8..+7, elements 8..15 = dims d0+16+half*8..+7.
    const int off = d0 + half * 8;
    float4 f0 = *(const float4*)(xrow + off);
    float4 f1 = *(const float4*)(xrow + off + 4);
    float4 f2 = *(const float4*)(xrow + off + 16);
    float4 f3 = *(const float4*)(xrow + off + 20);
    float va[16] = { f0.x, f0.y, f0.z, f0.w, f1.x, f1.y, f1.z, f1.w,
                     f2.x, f2.y, f2.z, f2.w, f3.x, f3.y, f3.z, f3.w };
    Frag ah, al;
#pragma unroll
    for (int i = 0; i < 16; ++i) {
      float v = va[i];
      xsq += v * v;
      _Float16 h = (_Float16)v;
      ah.h[i] = h;
      al.h[i] = (_Float16)(v - (float)h);
    }

#pragma unroll
    for (int t = 0; t < 16; ++t) {
      // B fragment: lane holds N column (centroid t*16+l15); elements 0..7 = dims d0+half*8..+7, 8..15 = dims d0+16+half*8..+7.
      Frag bh, bl;
#pragma unroll
      for (int i = 0; i < 16; ++i) {
        const int kk = off + (i < 8 ? i : 8 + i);
        bh.h[i] = ldsH[kk * CPAD + t * 16 + l15];
        bl.h[i] = ldsL[kk * CPAD + t * 16 + l15];
      }
      // x*c = h*h + h*l + l*h  (drop l*l: <= 2^-24 relative)
      acc[t] = __builtin_amdgcn_wmma_f32_16x16x32_f16(
          false, ah.v, false, bh.v, (short)0, acc[t], false, false);
      acc[t] = __builtin_amdgcn_wmma_f32_16x16x32_f16(
          false, ah.v, false, bl.v, (short)0, acc[t], false, false);
      acc[t] = __builtin_amdgcn_wmma_f32_16x16x32_f16(
          false, al.v, false, bh.v, (short)0, acc[t], false, false);
    }
  }

  // full ||x||^2 per row: combine the two lane-halves
  float xsq_full = xsq + __shfl_xor(xsq, 16, 32);
  if (half == 0) lds_xsq[wave * 16 + l15] = xsq_full;
  __syncthreads();

  // distances + argmin. C/D layout: VGPR j, lanes 0-15 -> (M=j, N=lane); lanes 16-31 -> (M=j+8, N=lane-16)
#pragma unroll
  for (int j = 0; j < 8; ++j) {
    const int M = j + 8 * half;
    const float xs = lds_xsq[wave * 16 + M];
    float best = 3.402823466e+38f;
    int bk = 0;
#pragma unroll
    for (int t = 0; t < 16; ++t) {
      int col = t * 16 + l15;
      float v = xs - 2.0f * acc[t][j] + lds_csq[col];
      if (v < best) { best = v; bk = col; }
    }
#pragma unroll
    for (int m = 1; m < 16; m <<= 1) {
      float ob = __shfl_xor(best, m, 32);
      int   ok = __shfl_xor(bk,   m, 32);
      if (ob < best || (ob == best && ok < bk)) { best = ob; bk = ok; }
    }
    if (l15 == 0) {
      int row = rowBase + M;
      assign_i[row] = bk;
      assign_f[row] = (float)bk;
      lds_minv[wave * 16 + M] = best;
    }
  }
  __syncthreads();

  // deterministic block reduction of the 128 per-row min distances
  for (int s = 64; s >= 1; s >>= 1) {
    if (tid < s) lds_minv[tid] += lds_minv[tid + s];
    __syncthreads();
  }
  if (tid == 0) blk_part[blockIdx.x] = lds_minv[0];
}

// ---------------------------------------------------------------- segment partial sums (no atomics)
__global__ __launch_bounds__(256, 1) void seg_kernel(
    const float* __restrict__ x, const int* __restrict__ assign_i,
    float* __restrict__ part_sums, float* __restrict__ part_cnts, int S) {
  __shared__ float sums[128 * DIMS];   // 128 clusters x 256 dims, 128KB
  __shared__ float cnts[128];
  __shared__ int achunk[256];

  const int s   = blockIdx.x >> 1;     // slice
  const int h   = blockIdx.x & 1;      // cluster half
  const int tid = threadIdx.x;
  const int kbase = h * 128;

  for (int i = tid; i < 128 * DIMS; i += 256) sums[i] = 0.f;
  if (tid < 128) cnts[tid] = 0.f;

  const int npb = NPTS / S;
  const int n0  = s * npb;
  for (int nb = 0; nb < npb; nb += 256) {
    __syncthreads();
    achunk[tid] = assign_i[n0 + nb + tid];
    __syncthreads();
    for (int i = 0; i < 256; ++i) {
      int a = achunk[i];                 // uniform across block
      if ((a >> 7) == h) {
        int n = n0 + nb + i;
        // thread tid exclusively owns dim column tid -> race-free plain add
        sums[(a - kbase) * DIMS + tid] += x[(size_t)n * DIMS + tid];
        if (tid == 0) cnts[a - kbase] += 1.f;
      }
    }
  }
  __syncthreads();
  for (int i = tid; i < 128 * DIMS; i += 256)
    part_sums[((size_t)s * NC + kbase) * DIMS + i] = sums[i];
  if (tid < 128) part_cnts[s * NC + kbase + tid] = cnts[tid];
}

// ---------------------------------------------------------------- means (empty cluster -> 0/0 = NaN)
__global__ void mean_kernel(const float* __restrict__ part_sums,
                            const float* __restrict__ part_cnts,
                            float* __restrict__ out_cent, int S) {
  const int k = blockIdx.x, d = threadIdx.x;
  __shared__ float csh;
  float acc = 0.f;
  for (int s = 0; s < S; ++s) acc += part_sums[((size_t)s * NC + k) * DIMS + d];
  if (d == 0) {
    float c = 0.f;
    for (int s = 0; s < S; ++s) c += part_cnts[s * NC + k];
    csh = c;
  }
  __syncthreads();
  out_cent[(size_t)k * DIMS + d] = acc / csh;
}

// ---------------------------------------------------------------- final deterministic distance sum
__global__ void finalsum_kernel(const float* __restrict__ blk_part, float* __restrict__ out_sum) {
  __shared__ float lds[256];
  const int tid = threadIdx.x;
  float a = 0.f;
  for (int i = tid; i < BLOCKS_ASSIGN; i += 256) a += blk_part[i];
  lds[tid] = a;
  __syncthreads();
  for (int s = 128; s >= 1; s >>= 1) {
    if (tid < s) lds[tid] += lds[tid + s];
    __syncthreads();
  }
  if (tid == 0) out_sum[0] = lds[0];
}

// ---------------------------------------------------------------- launch
extern "C" void kernel_launch(void* const* d_in, const int* in_sizes, int n_in,
                              void* d_out, int out_size, void* d_ws, size_t ws_size,
                              hipStream_t stream) {
  (void)in_sizes; (void)n_in; (void)out_size;
  const float* x    = (const float*)d_in[0];
  const float* cent = (const float*)d_in[1];

  float* out        = (float*)d_out;
  float* out_assign = out;                       // [N] assignments as float
  float* out_cent   = out + NPTS;                // [K*D]
  float* out_sum    = out + NPTS + NC * DIMS;    // [1]

  size_t off = 0;
  auto alloc = [&](size_t bytes) -> void* {
    void* p = (char*)d_ws + off;
    off += (bytes + 255) & ~(size_t)255;
    return p;
  };
  int*   assign_i = (int*)  alloc((size_t)NPTS * 4);
  float* blk_part = (float*)alloc((size_t)BLOCKS_ASSIGN * 4);
  float* csq      = (float*)alloc((size_t)NC * 4);

  // pick slice count S (power of two) that fits in ws; fixed per run -> deterministic
  int S = 64;
  while (S > 1 && off + (size_t)S * ((size_t)NC * 4 + (size_t)NC * DIMS * 4 + 512) > ws_size)
    S >>= 1;
  float* part_cnts = (float*)alloc((size_t)S * NC * 4);
  float* part_sums = (float*)alloc((size_t)S * NC * DIMS * 4);

  csq_kernel<<<1, 256, 0, stream>>>(cent, csq);
  assign_kernel<<<BLOCKS_ASSIGN, 256, 0, stream>>>(x, cent, csq, assign_i, out_assign, blk_part);
  seg_kernel<<<2 * S, 256, 0, stream>>>(x, assign_i, part_sums, part_cnts, S);
  mean_kernel<<<NC, DIMS, 0, stream>>>(part_sums, part_cnts, out_cent, S);
  finalsum_kernel<<<1, 256, 0, stream>>>(blk_part, out_sum);
}
